// GraphTransEncoderLayer_85186381349284
// MI455X (gfx1250) — matrix operand order repaired and checked by reading, not compiled, not run on hardware
//
#include <hip/hip_runtime.h>
#include <math.h>

#define DEV __device__ __forceinline__

typedef __bf16 bf16_t;
typedef __attribute__((ext_vector_type(16))) __bf16 v16bf;
typedef __attribute__((ext_vector_type(8)))  __bf16 v8bf;
typedef __attribute__((ext_vector_type(8)))  float  v8f;

constexpr int N_NODES = 50000;
constexpr int E_EDGES = 600000;

// ---------- bf16 helpers (explicit RNE, no reliance on cast codegen) ----------
DEV unsigned short f2bfbits(float f) {
  unsigned x = __float_as_uint(f);
  return (unsigned short)((x + 0x7FFFu + ((x >> 16) & 1u)) >> 16);
}
DEV bf16_t f2bf(float f) {
  unsigned short u = f2bfbits(f);
  return __builtin_bit_cast(bf16_t, u);
}
DEV float bfbits2f(unsigned short u) {
  return __uint_as_float(((unsigned)u) << 16);
}

// ---------- float atomic max via int/uint ordering trick ----------
DEV void atomicMaxF32(float* addr, float val) {
  int vi = __float_as_int(val);
  if (vi >= 0) atomicMax((int*)addr, vi);
  else         atomicMin((unsigned int*)addr, __float_as_uint(val));
}

// ---------- WMMA A-fragment loaders (16x32 bf16, ISA layout) ----------
// lane l: m = l&15, hi = l>>4; holds K = kbase..kbase+7 and kbase+16..kbase+23
// where kbase already includes hi*8.
DEV v16bf load_a_frag(const float* A, int lda, int row, int kbase) {
  const float* p = A + (size_t)row * lda + kbase;
  float4 f0 = *(const float4*)(p + 0);
  float4 f1 = *(const float4*)(p + 4);
  float4 f2 = *(const float4*)(p + 16);
  float4 f3 = *(const float4*)(p + 20);
  v16bf r;
  r[0] = f2bf(f0.x);  r[1] = f2bf(f0.y);  r[2] = f2bf(f0.z);  r[3] = f2bf(f0.w);
  r[4] = f2bf(f1.x);  r[5] = f2bf(f1.y);  r[6] = f2bf(f1.z);  r[7] = f2bf(f1.w);
  r[8] = f2bf(f2.x);  r[9] = f2bf(f2.y);  r[10] = f2bf(f2.z); r[11] = f2bf(f2.w);
  r[12] = f2bf(f3.x); r[13] = f2bf(f3.y); r[14] = f2bf(f3.z); r[15] = f2bf(f3.w);
  return r;
}
DEV v16bf load_a_frag(const bf16_t* A, int lda, int row, int kbase) {
  const bf16_t* p = A + (size_t)row * lda + kbase;
  v8bf a0 = *(const v8bf*)(p);
  v8bf a1 = *(const v8bf*)(p + 16);
  return __builtin_shufflevector(a0, a1, 0, 1, 2, 3, 4, 5, 6, 7,
                                         8, 9, 10, 11, 12, 13, 14, 15);
}

// ---------- WMMA GEMM:  C(M x KO) = A(M x K) * B(K x KO) [+epilogue] ----------
// MODE 0: out_f32 = acc + bias
// MODE 1: out_bf16 = acc                     (edge projection e)
// MODE 2: out_bf16 = gelu(acc + bias)        (FFN layer 1, exact erf GELU)
// MODE 3: v = acc + bias + res; out_f32 = v; accumulate sum/sumsq into stats
template <int K, int KO, int MODE, typename AT>
__global__ __launch_bounds__(256) void wmma_gemm_kernel(
    const AT* __restrict__ A, const bf16_t* __restrict__ B,
    const float* __restrict__ bias, const float* __restrict__ res,
    float* __restrict__ outf, bf16_t* __restrict__ outb,
    float* __restrict__ stats, int Mtiles) {
  const int wave = blockIdx.x * (blockDim.x >> 5) + (threadIdx.x >> 5);
  if (wave >= Mtiles) return;  // wave-granular: EXEC all ones at each WMMA
  const int lane = threadIdx.x & 31;
  const int hi = lane >> 4;
  const int l15 = lane & 15;
  const int m0 = wave << 4;
  constexpr int KF = K / 32;
  constexpr bool HOIST = (KF <= 4);
  float lsum = 0.f, lsq = 0.f;

  v16bf afr[HOIST ? KF : 1];
  if constexpr (HOIST) {
#pragma unroll
    for (int kf = 0; kf < KF; ++kf)
      afr[kf] = load_a_frag(A, K, m0 + l15, kf * 32 + hi * 8);
  }

  for (int nt = 0; nt < KO / 16; ++nt) {
    v8f acc = {0.f, 0.f, 0.f, 0.f, 0.f, 0.f, 0.f, 0.f};
#pragma unroll
    for (int kf = 0; kf < KF; ++kf) {
      v16bf a;
      if constexpr (HOIST) a = afr[kf];
      else                 a = load_a_frag(A, K, m0 + l15, kf * 32 + hi * 8);
      // B fragment: lane l holds col N = l&15, K = kf*32 + hi*8 + {0..7, 16..23}
      const bf16_t* bp = B + (size_t)(kf * 32 + hi * 8) * KO + nt * 16 + l15;
      v16bf b;
#pragma unroll
      for (int s = 0; s < 8; ++s) {
        b[s] = bp[(size_t)s * KO];
        b[s + 8] = bp[(size_t)(s + 16) * KO];
      }
      acc = __builtin_amdgcn_wmma_f32_16x16x32_bf16(
          false, a, false, b, (short)0, acc, false, false);
    }
    const int col = nt * 16 + l15;
#pragma unroll
    for (int j = 0; j < 8; ++j) {
      const int row = m0 + j + 8 * hi;  // D: VGPR j -> M=j (lanes 0-15), M=j+8 (16-31)
      const size_t idx = (size_t)row * KO + col;
      float v = acc[j];
      if constexpr (MODE == 0) {
        outf[idx] = v + bias[col];
      } else if constexpr (MODE == 1) {
        outb[idx] = f2bf(v);
      } else if constexpr (MODE == 2) {
        v += bias[col];
        float g = 0.5f * v * (1.0f + erff(v * 0.70710678118654752440f));
        outb[idx] = f2bf(g);
      } else {
        v += bias[col] + res[idx];
        outf[idx] = v;
        lsum += v;
        lsq += v * v;
      }
    }
  }
  if constexpr (MODE == 3) {
#pragma unroll
    for (int off = 16; off >= 1; off >>= 1) {
      lsum += __shfl_xor(lsum, off, 32);
      lsq += __shfl_xor(lsq, off, 32);
    }
    if (lane == 0) {
      atomicAdd(&stats[0], lsum);
      atomicAdd(&stats[1], lsq);
    }
  }
}

// ---------- weight packing ----------
__global__ __launch_bounds__(256) void pack_qkvr_kernel(
    const float* __restrict__ Wq, const float* __restrict__ Wk,
    const float* __restrict__ Wv, const float* __restrict__ Wskip,
    const float* __restrict__ bq, const float* __restrict__ bk,
    const float* __restrict__ bv, const float* __restrict__ bskip,
    bf16_t* __restrict__ Bp, float* __restrict__ biasp) {
  const int t = blockIdx.x * 256 + threadIdx.x;
  if (t >= 128 * 512) return;
  const int r = t >> 9, c = t & 511;
  const int sel = c >> 7, cc = c & 127;
  const float* W = sel == 0 ? Wq : sel == 1 ? Wk : sel == 2 ? Wv : Wskip;
  Bp[t] = f2bf(W[r * 128 + cc]);
  if (r == 0) {
    const float* bb = sel == 0 ? bq : sel == 1 ? bk : sel == 2 ? bv : bskip;
    biasp[c] = bb[cc];
  }
}

__global__ __launch_bounds__(256) void cvt_bf16_kernel(const float* __restrict__ in,
                                                       bf16_t* __restrict__ outp, int n) {
  const int t = blockIdx.x * 256 + threadIdx.x;
  if (t < n) outp[t] = f2bf(in[t]);
}

// ---------- attention scratch init ----------
__global__ __launch_bounds__(256) void init_attn_kernel(float* __restrict__ m,
                                                        float* __restrict__ s,
                                                        float* __restrict__ out_agg,
                                                        float* __restrict__ stats) {
  const size_t t = (size_t)blockIdx.x * 256 + threadIdx.x;
  if (t < (size_t)N_NODES * 128) out_agg[t] = 0.f;
  if (t < (size_t)N_NODES * 4) { m[t] = -INFINITY; s[t] = 0.f; }
  if (t < 8) stats[t] = 0.f;
}

// ---------- per-edge: alpha = q[dst].(k[src]+e)/sqrt(32), v_j = v[src]+e ----------
__global__ __launch_bounds__(256) void edge_alpha_kernel(
    const bf16_t* __restrict__ e_bf, const float* __restrict__ qkvr,
    const long long* __restrict__ ei, float* __restrict__ alpha,
    bf16_t* __restrict__ vj_bf) {
  const int eidx = blockIdx.x * 8 + (threadIdx.x >> 5);
  if (eidx >= E_EDGES) return;
  const int lane = threadIdx.x & 31;
  const int c0 = lane * 4;  // lane covers head = lane>>3
  const long long src = ei[eidx];
  const long long dst = ei[(size_t)E_EDGES + eidx];
  ushort4 eu = *(const ushort4*)((const unsigned short*)e_bf + (size_t)eidx * 128 + c0);
  const float e0 = bfbits2f(eu.x), e1 = bfbits2f(eu.y), e2 = bfbits2f(eu.z), e3 = bfbits2f(eu.w);
  float4 q4 = *(const float4*)(qkvr + (size_t)dst * 512 + c0);
  float4 k4 = *(const float4*)(qkvr + (size_t)src * 512 + 128 + c0);
  float4 v4 = *(const float4*)(qkvr + (size_t)src * 512 + 256 + c0);
  const float kj0 = k4.x + e0, kj1 = k4.y + e1, kj2 = k4.z + e2, kj3 = k4.w + e3;
  const float vj0 = v4.x + e0, vj1 = v4.y + e1, vj2 = v4.z + e2, vj3 = v4.w + e3;
  ushort4 vs = make_ushort4(f2bfbits(vj0), f2bfbits(vj1), f2bfbits(vj2), f2bfbits(vj3));
  *(ushort4*)((unsigned short*)vj_bf + (size_t)eidx * 128 + c0) = vs;
  float p = q4.x * kj0 + q4.y * kj1 + q4.z * kj2 + q4.w * kj3;
  p += __shfl_xor(p, 1, 32);
  p += __shfl_xor(p, 2, 32);
  p += __shfl_xor(p, 4, 32);  // 8 lanes per head
  if ((lane & 7) == 0)
    alpha[(size_t)eidx * 4 + (lane >> 3)] = p * 0.17677669529663688f;  // 1/sqrt(32)
}

// ---------- segment softmax ----------
__global__ __launch_bounds__(256) void seg_max_kernel(const float* __restrict__ alpha,
                                                      const long long* __restrict__ ei,
                                                      float* __restrict__ m) {
  const int t = blockIdx.x * 256 + threadIdx.x;
  if (t >= E_EDGES * 4) return;
  const int e = t >> 2, h = t & 3;
  const long long d = ei[(size_t)E_EDGES + e];
  atomicMaxF32(&m[d * 4 + h], alpha[t]);
}

__global__ __launch_bounds__(256) void exp_sum_kernel(float* __restrict__ alpha,
                                                      const long long* __restrict__ ei,
                                                      const float* __restrict__ m,
                                                      float* __restrict__ s) {
  const int t = blockIdx.x * 256 + threadIdx.x;
  if (t >= E_EDGES * 4) return;
  const int e = t >> 2, h = t & 3;
  const long long d = ei[(size_t)E_EDGES + e];
  const float a = expf(alpha[t] - m[d * 4 + h]);
  alpha[t] = a;
  atomicAdd(&s[d * 4 + h], a);
}

__global__ __launch_bounds__(256) void edge_agg_kernel(
    const float* __restrict__ alpha, const float* __restrict__ sbuf,
    const bf16_t* __restrict__ vj_bf, const long long* __restrict__ ei,
    float* __restrict__ out_agg) {
  const int eidx = blockIdx.x * 8 + (threadIdx.x >> 5);
  if (eidx >= E_EDGES) return;
  const int lane = threadIdx.x & 31;
  const int head = lane >> 3;
  const int c0 = lane * 4;
  const long long d = ei[(size_t)E_EDGES + eidx];
  const float coeff = alpha[(size_t)eidx * 4 + head] / (sbuf[d * 4 + head] + 1e-16f);
  ushort4 vu = *(const ushort4*)((const unsigned short*)vj_bf + (size_t)eidx * 128 + c0);
  float* op = out_agg + (size_t)d * 128 + c0;
  atomicAdd(op + 0, coeff * bfbits2f(vu.x));
  atomicAdd(op + 1, coeff * bfbits2f(vu.y));
  atomicAdd(op + 2, coeff * bfbits2f(vu.z));
  atomicAdd(op + 3, coeff * bfbits2f(vu.w));
}

// ---------- beta gate + residual + LN1 stats ----------
__global__ __launch_bounds__(256) void combine_kernel(
    const float* __restrict__ x, const float* __restrict__ out_agg,
    const float* __restrict__ qkvr, const float* __restrict__ Wbeta,
    float* __restrict__ y1, float* __restrict__ stats) {
  const int n = blockIdx.x * 8 + (threadIdx.x >> 5);
  if (n >= N_NODES) return;
  const int lane = threadIdx.x & 31;
  const int c0 = lane * 4;
  float4 o = *(const float4*)(out_agg + (size_t)n * 128 + c0);
  float4 xr = *(const float4*)(qkvr + (size_t)n * 512 + 384 + c0);
  float4 w0 = *(const float4*)(Wbeta + c0);
  float4 w1 = *(const float4*)(Wbeta + 128 + c0);
  float4 w2 = *(const float4*)(Wbeta + 256 + c0);
  float p = o.x * w0.x + o.y * w0.y + o.z * w0.z + o.w * w0.w
          + xr.x * w1.x + xr.y * w1.y + xr.z * w1.z + xr.w * w1.w
          + (o.x - xr.x) * w2.x + (o.y - xr.y) * w2.y
          + (o.z - xr.z) * w2.z + (o.w - xr.w) * w2.w;
  p += __shfl_xor(p, 1, 32);
  p += __shfl_xor(p, 2, 32);
  p += __shfl_xor(p, 4, 32);
  p += __shfl_xor(p, 8, 32);
  p += __shfl_xor(p, 16, 32);
  const float beta = 1.0f / (1.0f + expf(-p));
  float4 xv = *(const float4*)(x + (size_t)n * 128 + c0);
  const float r0 = xv.x + beta * xr.x + (1.0f - beta) * o.x;
  const float r1 = xv.y + beta * xr.y + (1.0f - beta) * o.y;
  const float r2 = xv.z + beta * xr.z + (1.0f - beta) * o.z;
  const float r3 = xv.w + beta * xr.w + (1.0f - beta) * o.w;
  *(float4*)(y1 + (size_t)n * 128 + c0) = make_float4(r0, r1, r2, r3);
  float lsum = r0 + r1 + r2 + r3;
  float lsq = r0 * r0 + r1 * r1 + r2 * r2 + r3 * r3;
#pragma unroll
  for (int off = 16; off >= 1; off >>= 1) {
    lsum += __shfl_xor(lsum, off, 32);
    lsq += __shfl_xor(lsq, off, 32);
  }
  if (lane == 0) {
    atomicAdd(&stats[0], lsum);
    atomicAdd(&stats[1], lsq);
  }
}

// ---------- global-scalar LayerNorm: finalize + apply ----------
__global__ void stats_finalize_kernel(float* __restrict__ stats, int base) {
  if (threadIdx.x == 0 && blockIdx.x == 0) {
    const float cnt = (float)N_NODES * 128.0f;
    const float mean = stats[base] / cnt;
    float var = stats[base + 1] / cnt - mean * mean;
    var = var > 0.f ? var : 0.f;
    stats[base + 2] = mean;
    stats[base + 3] = 1.0f / (sqrtf(var) + 1e-5f);
  }
}

__global__ __launch_bounds__(256) void ln_apply_kernel(
    const float* __restrict__ in, const float* __restrict__ stats, int base,
    const float* __restrict__ g, const float* __restrict__ b,
    float* __restrict__ outp) {
  const size_t t = (size_t)blockIdx.x * 256 + threadIdx.x;
  if (t >= (size_t)N_NODES * 128) return;
  const float mean = stats[base + 2];
  const float inv = stats[base + 3];
  const int c = (int)(t & 127);
  outp[t] = (in[t] - mean) * inv * g[c] + b[c];
}

// ============================== host ==============================
extern "C" void kernel_launch(void* const* d_in, const int* in_sizes, int n_in,
                              void* d_out, int out_size, void* d_ws, size_t ws_size,
                              hipStream_t stream) {
  (void)in_sizes; (void)n_in; (void)out_size; (void)ws_size;
  const float* x         = (const float*)d_in[0];
  const float* edge_attr = (const float*)d_in[1];
  const long long* ei    = (const long long*)d_in[2];  // int64 (2, E)
  const float* Wq    = (const float*)d_in[3];
  const float* bq    = (const float*)d_in[4];
  const float* Wk    = (const float*)d_in[5];
  const float* bk    = (const float*)d_in[6];
  const float* Wv    = (const float*)d_in[7];
  const float* bv    = (const float*)d_in[8];
  const float* We    = (const float*)d_in[9];
  const float* Wskip = (const float*)d_in[10];
  const float* bskip = (const float*)d_in[11];
  const float* Wbeta = (const float*)d_in[12];
  const float* g1    = (const float*)d_in[13];
  const float* be1   = (const float*)d_in[14];
  const float* g2    = (const float*)d_in[15];
  const float* be2   = (const float*)d_in[16];
  const float* W1    = (const float*)d_in[17];
  const float* bf1   = (const float*)d_in[18];
  const float* W2    = (const float*)d_in[19];
  const float* bf2   = (const float*)d_in[20];
  float* out = (float*)d_out;

  // workspace carving (~550 MB total)
  char* ws = (char*)d_ws;
  size_t off = 0;
  auto alloc = [&](size_t bytes) -> char* {
    char* p = ws + off;
    off += (bytes + 255) & ~(size_t)255;
    return p;
  };
  float*  qkvr    = (float*)alloc((size_t)N_NODES * 512 * 4);   // [q|k|v|x_r]
  bf16_t* e_bf    = (bf16_t*)alloc((size_t)E_EDGES * 128 * 2);
  bf16_t* vj_bf   = (bf16_t*)alloc((size_t)E_EDGES * 128 * 2);
  float*  alpha   = (float*)alloc((size_t)E_EDGES * 4 * 4);
  float*  mbuf    = (float*)alloc((size_t)N_NODES * 4 * 4);
  float*  sbuf    = (float*)alloc((size_t)N_NODES * 4 * 4);
  float*  out_agg = (float*)alloc((size_t)N_NODES * 128 * 4);
  float*  y1buf   = (float*)alloc((size_t)N_NODES * 128 * 4);
  float*  x1buf   = (float*)alloc((size_t)N_NODES * 128 * 4);
  bf16_t* h_bf    = (bf16_t*)alloc((size_t)N_NODES * 512 * 2);
  bf16_t* Bqkvr   = (bf16_t*)alloc(128 * 512 * 2);
  float*  biasq   = (float*)alloc(512 * 4);
  bf16_t* We_bf   = (bf16_t*)alloc(128 * 128 * 2);
  bf16_t* W1_bf   = (bf16_t*)alloc(128 * 512 * 2);
  bf16_t* W2_bf   = (bf16_t*)alloc(512 * 128 * 2);
  float*  stats   = (float*)alloc(32);  // [sum1,sq1,mean1,inv1, sum2,sq2,mean2,inv2]

  // weight prep
  pack_qkvr_kernel<<<256, 256, 0, stream>>>(Wq, Wk, Wv, Wskip, bq, bk, bv, bskip,
                                            Bqkvr, biasq);
  cvt_bf16_kernel<<<64, 256, 0, stream>>>(We, We_bf, 128 * 128);
  cvt_bf16_kernel<<<256, 256, 0, stream>>>(W1, W1_bf, 128 * 512);
  cvt_bf16_kernel<<<256, 256, 0, stream>>>(W2, W2_bf, 512 * 128);
  init_attn_kernel<<<25000, 256, 0, stream>>>(mbuf, sbuf, out_agg, stats);

  // fused q|k|v|skip projection: (50000x128) @ (128x512)
  wmma_gemm_kernel<128, 512, 0, float><<<391, 256, 0, stream>>>(
      x, Bqkvr, biasq, nullptr, qkvr, nullptr, nullptr, 3125);
  // edge projection: (600000x128) @ (128x128) -> bf16
  wmma_gemm_kernel<128, 128, 1, float><<<4688, 256, 0, stream>>>(
      edge_attr, We_bf, nullptr, nullptr, nullptr, e_bf, nullptr, 37500);

  // attention
  edge_alpha_kernel<<<75000, 256, 0, stream>>>(e_bf, qkvr, ei, alpha, vj_bf);
  seg_max_kernel<<<9375, 256, 0, stream>>>(alpha, ei, mbuf);
  exp_sum_kernel<<<9375, 256, 0, stream>>>(alpha, ei, mbuf, sbuf);
  edge_agg_kernel<<<75000, 256, 0, stream>>>(alpha, sbuf, vj_bf, ei, out_agg);

  // gate + residual + LN1
  combine_kernel<<<6250, 256, 0, stream>>>(x, out_agg, qkvr, Wbeta, y1buf, stats);
  stats_finalize_kernel<<<1, 32, 0, stream>>>(stats, 0);
  ln_apply_kernel<<<25000, 256, 0, stream>>>(y1buf, stats, 0, g1, be1, x1buf);

  // FFN: (50000x128)@(128x512) GELU -> bf16, then (50000x512)@(512x128) + residual
  wmma_gemm_kernel<128, 512, 2, float><<<391, 256, 0, stream>>>(
      x1buf, W1_bf, bf1, nullptr, nullptr, h_bf, nullptr, 3125);
  wmma_gemm_kernel<512, 128, 3, bf16_t><<<391, 256, 0, stream>>>(
      h_bf, W2_bf, bf2, x1buf, out, nullptr, stats + 4, 3125);
  stats_finalize_kernel<<<1, 32, 0, stream>>>(stats, 4);
  ln_apply_kernel<<<25000, 256, 0, stream>>>(out, stats, 4, g2, be2, out);
}
